// LayerWiseBaseEpoch_13700945674793
// MI455X (gfx1250) — compile-verified, matrix operand order checked by reading, not run on hardware
//
#include <hip/hip_runtime.h>
#include <hip/hip_bf16.h>
#include <math.h>

typedef __attribute__((ext_vector_type(16))) __bf16 v16bf;
typedef __attribute__((ext_vector_type(8)))  __bf16 v8bf;
typedef __attribute__((ext_vector_type(8)))  float  v8f;
typedef unsigned short ushort_t;

#define Bsz   32
#define Lnum  13
#define Tlen  500
#define Ddim  768
#define ROWS  (Bsz * Tlen)      // 16000 rows (b,t) per layer
#define PITCH 776               // bf16 elements per LDS row (768 + 8 pad, 16B aligned)
#define KT    24                // 768 / 32
#define NT    48                // 768 / 16
#define KBATCH 4                // kt per load batch (8 b128 loads in flight)

__device__ __forceinline__ ushort_t f2bf(float x) {
    __bf16 h = (__bf16)x;                 // RNE f32->bf16
    return __builtin_bit_cast(ushort_t, h);
}

// branch-free tanh: 1 - 2/(exp(2x)+1); saturates correctly, ~5 VALU + v_exp_f32
__device__ __forceinline__ float fast_tanh(float x) {
    float e = __expf(2.0f * x);
    return 1.0f - 2.0f * __builtin_amdgcn_rcpf(e + 1.0f);
}

// ---------------- Kernel 0: W[l][d][e] f32 -> Wt[l][e][d] bf16 (tiled transpose) --------
__global__ void wt_transpose_kernel(const float* __restrict__ W, ushort_t* __restrict__ Wt) {
    __shared__ float tile[32][33];
    int l = blockIdx.z;
    int ebase = blockIdx.x * 32;
    int dbase = blockIdx.y * 32;
    const float* Wl  = W  + (size_t)l * Ddim * Ddim;
    ushort_t*    Wtl = Wt + (size_t)l * Ddim * Ddim;
    int tx = threadIdx.x;                                   // 0..31
    for (int ty = threadIdx.y; ty < 32; ty += 8)
        tile[ty][tx] = Wl[(size_t)(dbase + ty) * Ddim + (ebase + tx)];
    __syncthreads();
    for (int ty = threadIdx.y; ty < 32; ty += 8)
        Wtl[(size_t)(ebase + ty) * Ddim + (dbase + tx)] = f2bf(tile[tx][ty]);
}

// ---------------- Kernel 1: scores[l][row] = v_l . tanh(x W_l + b_l) + c_l --------------
// One wave per block; wave owns TWO 16-row M-tiles (32 rows) so each B-fragment
// feeds two WMMAs (halves W_l L2 traffic). B-fragments are loaded in batches of
// KBATCH kt into distinct register buffers before any consumption, so the waits
// decay loadcnt<=6/4/2/0 across the batch instead of a full stall per WMMA pair.
__global__ void __launch_bounds__(32)
score_kernel(const float* __restrict__ x, const ushort_t* __restrict__ Wt,
             const float* __restrict__ bias, const float* __restrict__ vvec,
             const float* __restrict__ cvec, float* __restrict__ scores) {
    __shared__ ushort_t smem[32 * PITCH];                   // 32 rows, ~49.7 KB
    int l        = blockIdx.x / 500;
    int rowBase  = (blockIdx.x % 500) * 32;
    int lane     = threadIdx.x & 31;

    // Stage 32 A-rows (f32 global -> bf16 LDS)
    for (int r = 0; r < 32; ++r) {
        int g  = rowBase + r;
        int bb = g / Tlen, tt = g - bb * Tlen;
        const float* xr = x + (((size_t)bb * Lnum + l) * Tlen + tt) * Ddim;
        ushort_t* dst = smem + r * PITCH;
        for (int i = lane; i < Ddim / 4; i += 32) {
            float4 f = *(const float4*)(xr + i * 4);
            ushort_t p[4] = { f2bf(f.x), f2bf(f.y), f2bf(f.z), f2bf(f.w) };
            *(unsigned long long*)(dst + i * 4) = *(const unsigned long long*)p;
        }
    }
    __syncthreads();

    int half = lane >> 4;                 // K sub-block selector (ISA bf16 A/B layout)
    int mn   = lane & 15;                 // A: row M; B/C/D: column N
    const ushort_t* a0p = smem + mn * PITCH + half * 8;         // M-tile 0
    const ushort_t* a1p = smem + (16 + mn) * PITCH + half * 8;  // M-tile 1
    const ushort_t* WtL = Wt + (size_t)l * Ddim * Ddim;

    float rowAcc0[8], rowAcc1[8];         // per-lane partial row sums (deferred reduce)
#pragma unroll
    for (int j = 0; j < 8; ++j) { rowAcc0[j] = 0.f; rowAcc1[j] = 0.f; }

    for (int nt = 0; nt < NT; ++nt) {
        int e = nt * 16 + mn;
        float bval = bias[l * Ddim + e];
        float vval = vvec[l * Ddim + e];
        const ushort_t* b0p = WtL + (size_t)e * Ddim + half * 8;
        __builtin_prefetch((const void*)(b0p + 16 * (size_t)Ddim), 0, 3); // next N-tile

        v8f acc0 = {}, acc1 = {};
        for (int kb = 0; kb < KT; kb += KBATCH) {
            // Phase 1: issue all B loads of this batch into distinct buffers
            v8bf bb[2 * KBATCH];
#pragma unroll
            for (int q = 0; q < KBATCH; ++q) {
                bb[2 * q]     = *(const v8bf*)(b0p + (kb + q) * 32);
                bb[2 * q + 1] = *(const v8bf*)(b0p + (kb + q) * 32 + 16);
            }
            // Phase 2: consume in order (waits decay across the batch)
#pragma unroll
            for (int q = 0; q < KBATCH; ++q) {
                int kt = kb + q;
                v8bf a00 = *(const v8bf*)(a0p + kt * 32);
                v8bf a01 = *(const v8bf*)(a0p + kt * 32 + 16);
                v8bf a10 = *(const v8bf*)(a1p + kt * 32);
                v8bf a11 = *(const v8bf*)(a1p + kt * 32 + 16);
                v16bf Bm = __builtin_shufflevector(bb[2 * q], bb[2 * q + 1],
                                                   0,1,2,3,4,5,6,7,8,9,10,11,12,13,14,15);
                v16bf A0 = __builtin_shufflevector(a00, a01,
                                                   0,1,2,3,4,5,6,7,8,9,10,11,12,13,14,15);
                v16bf A1 = __builtin_shufflevector(a10, a11,
                                                   0,1,2,3,4,5,6,7,8,9,10,11,12,13,14,15);
                acc0 = __builtin_amdgcn_wmma_f32_16x16x32_bf16(
                    false, A0, false, Bm, (short)0, acc0, false, false);
                acc1 = __builtin_amdgcn_wmma_f32_16x16x32_bf16(
                    false, A1, false, Bm, (short)0, acc1, false, false);
            }
        }
        // epilogue: branch-free tanh(h+b)*v; cross-lane reduce deferred to end
#pragma unroll
        for (int j = 0; j < 8; ++j) {
            rowAcc0[j] += fast_tanh(acc0[j] + bval) * vval;
            rowAcc1[j] += fast_tanh(acc1[j] + bval) * vval;
        }
    }

    // one 4-step xor-shuffle reduction per row group; lanes 0/16 hold both halves
    float cval = cvec[l];
#pragma unroll
    for (int j = 0; j < 8; ++j) {
        float t0 = rowAcc0[j], t1 = rowAcc1[j];
        t0 += __shfl_xor(t0, 1, 32); t1 += __shfl_xor(t1, 1, 32);
        t0 += __shfl_xor(t0, 2, 32); t1 += __shfl_xor(t1, 2, 32);
        t0 += __shfl_xor(t0, 4, 32); t1 += __shfl_xor(t1, 4, 32);
        t0 += __shfl_xor(t0, 8, 32); t1 += __shfl_xor(t1, 8, 32);
        rowAcc0[j] = t0; rowAcc1[j] = t1;
    }
    if (lane == 0 || lane == 16) {        // lane0: rows j of half0; lane16: rows 8+j
        int rbase = rowBase + half * 8;
#pragma unroll
        for (int j = 0; j < 8; ++j) {
            scores[(size_t)l * ROWS + rbase + j]      = rowAcc0[j] + cval;
            scores[(size_t)l * ROWS + rbase + 16 + j] = rowAcc1[j] + cval;
        }
    }
}

// ---------------- Kernel 2: stable ranks, mask bottom-k, softmax over T ----------------
__global__ void softmax_mask_kernel(const float* __restrict__ scores, float* __restrict__ attn) {
    __shared__ float ls[Tlen];
    __shared__ float ms[Tlen];
    __shared__ float red[256];
    int bl = blockIdx.x;                  // 0..B*L-1
    int b = bl / Lnum, l = bl % Lnum;
    const float* s = scores + (size_t)l * ROWS + (size_t)b * Tlen;
    int tid = threadIdx.x;
    for (int i = tid; i < Tlen; i += 256) ls[i] = s[i];
    __syncthreads();
    const int kmask = Tlen / 2;           // 250
    for (int i = tid; i < Tlen; i += 256) {
        float si = ls[i];
        int rank = 0;
        for (int j = 0; j < Tlen; ++j) {
            float sj = ls[j];
            rank += (sj < si) || (sj == si && j < i);   // stable ascending rank
        }
        ms[i] = (rank < kmask) ? -INFINITY : si;
    }
    __syncthreads();
    float m = -INFINITY;
    for (int i = tid; i < Tlen; i += 256) m = fmaxf(m, ms[i]);
    red[tid] = m; __syncthreads();
    for (int st = 128; st > 0; st >>= 1) { if (tid < st) red[tid] = fmaxf(red[tid], red[tid + st]); __syncthreads(); }
    m = red[0]; __syncthreads();
    float psum = 0.f;
    for (int i = tid; i < Tlen; i += 256) { float e = __expf(ms[i] - m); ms[i] = e; psum += e; }
    red[tid] = psum; __syncthreads();
    for (int st = 128; st > 0; st >>= 1) { if (tid < st) red[tid] += red[tid + st]; __syncthreads(); }
    float inv = 1.f / red[0];
    float* ab = attn + ((size_t)b * Lnum + l) * Tlen;
    for (int i = tid; i < Tlen; i += 256) ab[i] = ms[i] * inv;
}

// ---------------- Kernel 3: out[b,d] = (1/T) sum_{l,t} w_l*attn[b,l,t]*x[b,l,t,d] -------
__global__ void wsum_kernel(const float* __restrict__ x, const float* __restrict__ attn,
                            const float* __restrict__ lw, float* __restrict__ out) {
    const int TCH = 25;                   // t per chunk; 20 chunks cover T=500
    __shared__ float coef[Lnum * TCH];
    int b  = blockIdx.x / 20;
    int t0 = (blockIdx.x % 20) * TCH;
    int tid = threadIdx.x;
    for (int i = tid; i < Lnum * TCH; i += 256) {
        int l = i / TCH, tt = i - l * TCH;
        coef[i] = lw[l] * attn[((size_t)b * Lnum + l) * Tlen + t0 + tt];
    }
    __syncthreads();
    float acc0 = 0.f, acc1 = 0.f, acc2 = 0.f;
    for (int l = 0; l < Lnum; ++l) {
        const float* xb = x + (((size_t)b * Lnum + l) * Tlen + t0) * Ddim;
        for (int tt = 0; tt < TCH; ++tt) {
            float cf = coef[l * TCH + tt];
            const float* xr = xb + (size_t)tt * Ddim;
            acc0 += cf * xr[tid];
            acc1 += cf * xr[tid + 256];
            acc2 += cf * xr[tid + 512];
        }
    }
    const float s = 1.f / (float)Tlen;
    atomicAdd(out + (size_t)b * Ddim + tid,       acc0 * s);
    atomicAdd(out + (size_t)b * Ddim + tid + 256, acc1 * s);
    atomicAdd(out + (size_t)b * Ddim + tid + 512, acc2 * s);
}

// ---------------- Launch ---------------------------------------------------------------
extern "C" void kernel_launch(void* const* d_in, const int* in_sizes, int n_in,
                              void* d_out, int out_size, void* d_ws, size_t ws_size,
                              hipStream_t stream) {
    const float* x    = (const float*)d_in[0];   // [B,L,T,D]
    const float* W    = (const float*)d_in[1];   // [L,D,D]
    const float* bias = (const float*)d_in[2];   // [L,D]
    const float* vvec = (const float*)d_in[3];   // [L,D]
    const float* cvec = (const float*)d_in[4];   // [L]
    const float* lw   = (const float*)d_in[5];   // [L]
    float* out = (float*)d_out;
    (void)in_sizes; (void)n_in; (void)ws_size;

    char* ws = (char*)d_ws;
    size_t off = 0;
    ushort_t* Wt = (ushort_t*)(ws + off);                       // bf16 [L][e][d]
    off += (size_t)Lnum * Ddim * Ddim * sizeof(ushort_t);
    off  = (off + 255) & ~(size_t)255;
    float* scores = (float*)(ws + off);                         // f32 [L][ROWS]
    off += (size_t)Lnum * ROWS * sizeof(float);
    off  = (off + 255) & ~(size_t)255;
    float* attn = (float*)(ws + off);                           // f32 [B][L][T]

    (void)hipMemsetAsync(d_out, 0, (size_t)out_size * sizeof(float), stream);
    wt_transpose_kernel<<<dim3(Ddim / 32, Ddim / 32, Lnum), dim3(32, 8), 0, stream>>>(W, Wt);
    score_kernel<<<dim3(Lnum * 500), dim3(32), 0, stream>>>(x, Wt, bias, vvec, cvec, scores);
    softmax_mask_kernel<<<dim3(Bsz * Lnum), dim3(256), 0, stream>>>(scores, attn);
    wsum_kernel<<<dim3(Bsz * 20), dim3(256), 0, stream>>>(x, attn, lw, out);
}
